// PointNet2Classifier_26998164422713
// MI455X (gfx1250) — matrix operand order repaired and checked by reading, not compiled, not run on hardware
//
#include <hip/hip_runtime.h>
#include <hip/hip_bf16.h>
#include <stdint.h>

// ---------------------------------------------------------------------------
// Types for WMMA fragments (wave32, v_wmma_f32_16x16x32_f16)
// ---------------------------------------------------------------------------
typedef __attribute__((ext_vector_type(16))) _Float16 v16h;
typedef __attribute__((ext_vector_type(8)))  _Float16 v8h;
typedef __attribute__((ext_vector_type(8)))  float    v8f;

#define FLT_BIG 3.4e38f

// ---------------------------------------------------------------------------
// Weight pad + fp32 -> f16 convert:  wp[CoutPad][CinPad], zero padded
// ---------------------------------------------------------------------------
__global__ void pad_weights_kernel(const float* __restrict__ w, _Float16* __restrict__ wp,
                                   int Cout, int Cin, int CoutPad, int CinPad) {
  long t = (long)blockIdx.x * blockDim.x + threadIdx.x;
  long total = (long)CoutPad * CinPad;
  if (t >= total) return;
  int r = (int)(t / CinPad);
  int c = (int)(t % CinPad);
  float v = (r < Cout && c < Cin) ? w[(long)r * Cin + c] : 0.0f;
  wp[t] = (_Float16)v;
}

// ---------------------------------------------------------------------------
// kNN: one block per query (b,p). d2 table in LDS; per-round argmin via
// wave32 shuffle reduction (2 barriers/round instead of ~9).
// x layout: x[b][ch][n], ch stride 4096, batch stride 6*4096.
// ---------------------------------------------------------------------------
__global__ void knn_kernel(const float* __restrict__ x, int P, int Npts, int K,
                           int* __restrict__ idx) {
  extern __shared__ float smem[];
  float* d2s  = smem;                      // [Npts]
  float* wval = smem + Npts;               // [nwaves]
  int*   wind = (int*)(wval + 8);          // [nwaves]

  int b = blockIdx.x / P;
  int p = blockIdx.x % P;
  const float* xb = x + (size_t)b * 6 * 4096;
  float qx = xb[p], qy = xb[4096 + p], qz = xb[8192 + p];

  for (int n = threadIdx.x; n < Npts; n += blockDim.x) {
    float dx = xb[n] - qx, dy = xb[4096 + n] - qy, dz = xb[8192 + n] - qz;
    d2s[n] = dx * dx + dy * dy + dz * dz;
  }
  __syncthreads();

  int nwaves = blockDim.x >> 5;
  int wave = threadIdx.x >> 5, lane = threadIdx.x & 31;
  int* out = idx + (size_t)blockIdx.x * K;

  for (int kk = 0; kk < K; ++kk) {
    float best = FLT_BIG; int bi = 0;
    for (int n = threadIdx.x; n < Npts; n += blockDim.x) {
      float v = d2s[n];
      if (v < best) { best = v; bi = n; }
    }
    // wave32 shuffle reduction
#pragma unroll
    for (int off = 16; off > 0; off >>= 1) {
      float ov = __shfl_down(best, off, 32);
      int   oi = __shfl_down(bi,   off, 32);
      if (ov < best) { best = ov; bi = oi; }
    }
    if (lane == 0) { wval[wave] = best; wind[wave] = bi; }
    __syncthreads();
    if (threadIdx.x == 0) {
      float bv = wval[0]; int sel = wind[0];
      for (int w = 1; w < nwaves; ++w)
        if (wval[w] < bv) { bv = wval[w]; sel = wind[w]; }
      out[kk] = sel;
      d2s[sel] = FLT_BIG;
    }
    __syncthreads();
  }
}

// ---------------------------------------------------------------------------
// SA1 grouping: feat column (b,p,kk) = [xyz_i - xyz_p (3), pts_i (3), pad->32]
// ---------------------------------------------------------------------------
__global__ void gather_sa1_kernel(const float* __restrict__ x, const int* __restrict__ idx,
                                  _Float16* __restrict__ feat) {
  long n = (long)blockIdx.x * blockDim.x + threadIdx.x;
  const long total = 32L * 512 * 32;
  if (n >= total) return;
  long pq = n >> 5;                 // (b*512+p)
  int p = (int)(pq & 511);
  int b = (int)(pq >> 9);
  const float* xb = x + (size_t)b * 6 * 4096;
  int i = idx[n];
  _Float16* f = feat + n * 32;
#pragma unroll
  for (int c = 0; c < 3; ++c)
    f[c] = (_Float16)(xb[c * 4096 + i] - xb[c * 4096 + p]);
#pragma unroll
  for (int c = 0; c < 3; ++c)
    f[3 + c] = (_Float16)(xb[(3 + c) * 4096 + i]);
#pragma unroll
  for (int c = 6; c < 32; ++c) f[c] = (_Float16)0.0f;
}

// ---------------------------------------------------------------------------
// SA2 grouping: feat column (b,p,kk) = [xyz_i - xyz_p (3), p1[i] (128), pad->160]
// ---------------------------------------------------------------------------
__global__ void gather_sa2_kernel(const float* __restrict__ x, const int* __restrict__ idx,
                                  const _Float16* __restrict__ p1, _Float16* __restrict__ feat) {
  long n = (long)blockIdx.x * blockDim.x + threadIdx.x;
  const long total = 32L * 128 * 64;
  if (n >= total) return;
  long pq = n >> 6;                 // (b*128+p)
  int p = (int)(pq & 127);
  int b = (int)(pq >> 7);
  const float* xb = x + (size_t)b * 6 * 4096;
  int i = idx[n];
  _Float16* f = feat + n * 160;
#pragma unroll
  for (int c = 0; c < 3; ++c)
    f[c] = (_Float16)(xb[c * 4096 + i] - xb[c * 4096 + p]);
  const _Float16* src = p1 + ((size_t)b * 512 + i) * 128;
  for (int c = 0; c < 128; ++c) f[3 + c] = src[c];
#pragma unroll
  for (int c = 131; c < 160; ++c) f[c] = (_Float16)0.0f;
}

// ---------------------------------------------------------------------------
// SA3 grouping: nsample=1 -> grouped_xyz==0; feat = [0,0,0, p2 (256), pad->288]
// ---------------------------------------------------------------------------
__global__ void gather_sa3_kernel(const _Float16* __restrict__ p2, _Float16* __restrict__ feat) {
  int q = blockIdx.x * blockDim.x + threadIdx.x;   // b*128+p, 4096 cols
  if (q >= 32 * 128) return;
  _Float16* f = feat + (size_t)q * 288;
  f[0] = f[1] = f[2] = (_Float16)0.0f;
  const _Float16* src = p2 + (size_t)q * 256;
  for (int c = 0; c < 256; ++c) f[3 + c] = src[c];
#pragma unroll
  for (int c = 259; c < 288; ++c) f[c] = (_Float16)0.0f;
}

// ---------------------------------------------------------------------------
// Register-blocked WMMA GEMM: Y[n][m] = act(sum_k W[m][k] * X[n][k] + b[m])
//  - macro-tile per wave: 32 rows (2 M-tiles) x 64 cols (4 N-tiles)
//  - 8 accumulators; each K-step: 2 A-fragments + 4 B-fragments -> 8 WMMAs
//    (A reused 4x, B reused 2x vs. 1 WMMA / 2 fragments before)
//  - tile index readfirstlane'd -> scalar guard, EXEC all-ones around WMMA
//  - requires CoutPad % 32 == 0, NcolsPad % 64 == 0 (host guarantees)
// ---------------------------------------------------------------------------
__global__ void wmma_gemm_kernel(const _Float16* __restrict__ W, const _Float16* __restrict__ X,
                                 const float* __restrict__ bias, int Cout, int CinPad,
                                 int mt2n, long macros, long Ncols,
                                 _Float16* __restrict__ Yh, float* __restrict__ Yf,
                                 int ldy, int relu) {
  int wave = __builtin_amdgcn_readfirstlane((int)(threadIdx.x >> 5));
  long id = (long)blockIdx.x * (blockDim.x >> 5) + wave;
  if (id >= macros) return;                        // scalar branch (uniform)
  int  mt2 = (int)(id % mt2n);
  long nt4 = id / mt2n;

  int lane = threadIdx.x & 31;
  int lm = lane & 15, lh = lane >> 4;
  int aksel = lh * 16;                             // A: K half per half-wave
  int bksel = lh * 16;                             // B: K half per half-wave

  const _Float16* wrow0 = W + (size_t)(mt2 * 32 + lm) * CinPad;
  const _Float16* wrow1 = wrow0 + (size_t)16 * CinPad;
  long n0 = nt4 * 64;
  const _Float16* xc0 = X + (size_t)(n0 + lm) * CinPad + bksel;
  size_t xstep = (size_t)16 * CinPad;

  v8f c[2][4];
#pragma unroll
  for (int i = 0; i < 2; ++i)
#pragma unroll
    for (int j = 0; j < 4; ++j) c[i][j] = (v8f){};

  for (int k0 = 0; k0 < CinPad; k0 += 32) {
    // A fragments: 16 contiguous K per row (matches B)
    v16h a0 = *(const v16h*)(wrow0 + k0 + aksel);
    v16h a1 = *(const v16h*)(wrow1 + k0 + aksel);
    // B fragments: 16 contiguous K per column
    v16h b0 = *(const v16h*)(xc0 + k0);
    v16h b1 = *(const v16h*)(xc0 + xstep + k0);
    v16h b2 = *(const v16h*)(xc0 + 2 * xstep + k0);
    v16h b3 = *(const v16h*)(xc0 + 3 * xstep + k0);
    __builtin_prefetch(xc0 + k0 + 32, 0, 1);               // global_prefetch_b8
    __builtin_prefetch(xc0 + 2 * xstep + k0 + 32, 0, 1);

    c[0][0] = __builtin_amdgcn_wmma_f32_16x16x32_f16(false, a0, false, b0, (short)0, c[0][0], false, false);
    c[0][1] = __builtin_amdgcn_wmma_f32_16x16x32_f16(false, a0, false, b1, (short)0, c[0][1], false, false);
    c[0][2] = __builtin_amdgcn_wmma_f32_16x16x32_f16(false, a0, false, b2, (short)0, c[0][2], false, false);
    c[0][3] = __builtin_amdgcn_wmma_f32_16x16x32_f16(false, a0, false, b3, (short)0, c[0][3], false, false);
    c[1][0] = __builtin_amdgcn_wmma_f32_16x16x32_f16(false, a1, false, b0, (short)0, c[1][0], false, false);
    c[1][1] = __builtin_amdgcn_wmma_f32_16x16x32_f16(false, a1, false, b1, (short)0, c[1][1], false, false);
    c[1][2] = __builtin_amdgcn_wmma_f32_16x16x32_f16(false, a1, false, b2, (short)0, c[1][2], false, false);
    c[1][3] = __builtin_amdgcn_wmma_f32_16x16x32_f16(false, a1, false, b3, (short)0, c[1][3], false, false);
  }

  // Epilogue: C layout -> VGPR r, lanes 16-31 hold M+8; N = lane%16
#pragma unroll
  for (int i = 0; i < 2; ++i) {
    int mbase = mt2 * 32 + i * 16 + lh * 8;
#pragma unroll
    for (int j = 0; j < 4; ++j) {
      long n = n0 + j * 16 + lm;
#pragma unroll
      for (int r = 0; r < 8; ++r) {
        int m = mbase + r;
        if (m < Cout && n < Ncols) {
          float v = c[i][j][r] + bias[m];
          if (relu) v = fmaxf(v, 0.0f);
          size_t off = (size_t)n * ldy + m;
          if (Yh) Yh[off] = (_Float16)v;
          else    Yf[off] = v;
        }
      }
    }
  }
}

// ---------------------------------------------------------------------------
// Max-pool over the nsample (K) axis: out[q][c] = max_kk in[(q*K+kk)][c]
// ---------------------------------------------------------------------------
__global__ void maxpool_k_kernel(const _Float16* __restrict__ in, _Float16* __restrict__ out,
                                 int K, int C, long Q) {
  long t = (long)blockIdx.x * blockDim.x + threadIdx.x;
  if (t >= Q * C) return;
  long q = t / C;
  int  c = (int)(t % C);
  const _Float16* src = in + (size_t)q * K * C + c;
  float m = -FLT_BIG;
  for (int kk = 0; kk < K; ++kk) m = fmaxf(m, (float)src[(size_t)kk * C]);
  out[t] = (_Float16)m;
}

// ---------------------------------------------------------------------------
// Global max-pool over points for SA3: g[b][c] = max_p p3[(b*128+p)][c]
// ---------------------------------------------------------------------------
__global__ void pool_g_kernel(const _Float16* __restrict__ p3, _Float16* __restrict__ g) {
  int t = blockIdx.x * blockDim.x + threadIdx.x;   // 32*1024
  if (t >= 32 * 1024) return;
  int b = t >> 10, c = t & 1023;
  const _Float16* src = p3 + ((size_t)b * 128) * 1024 + c;
  float m = -FLT_BIG;
  for (int p = 0; p < 128; ++p) m = fmaxf(m, (float)src[(size_t)p * 1024]);
  g[(size_t)b * 1024 + c] = (_Float16)m;
}

// ---------------------------------------------------------------------------
// Host-side orchestration
// ---------------------------------------------------------------------------
static inline long cdiv(long a, long b) { return (a + b - 1) / b; }

extern "C" void kernel_launch(void* const* d_in, const int* in_sizes, int n_in,
                              void* d_out, int out_size, void* d_ws, size_t ws_size,
                              hipStream_t stream) {
  (void)in_sizes; (void)n_in; (void)out_size; (void)ws_size;
  const float* x = (const float*)d_in[0];

  // ----- workspace carve-out ------------------------------------------------
  char* ws = (char*)d_ws;
  size_t cur = 0;
  auto alloc = [&](size_t bytes) -> void* {
    void* p = ws + cur;
    cur = (cur + bytes + 255) & ~(size_t)255;
    return p;
  };

  // Padded f16 weights: {input idx, Cout, Cin, CoutPad(%32), CinPad(%32)}
  struct WD { int ii, Cout, Cin, CoutPad, CinPad; _Float16* wp; };
  WD wd[12] = {
    {1,   64,    6,   64,   32, nullptr},  // sa1_w1
    {3,   64,   64,   64,   64, nullptr},  // sa1_w2
    {5,  128,   64,  128,   64, nullptr},  // sa1_w3
    {7,  128,  131,  128,  160, nullptr},  // sa2_w1
    {9,  128,  128,  128,  128, nullptr},  // sa2_w2
    {11, 256,  128,  256,  128, nullptr},  // sa2_w3
    {13, 256,  259,  256,  288, nullptr},  // sa3_w1
    {15, 512,  256,  512,  256, nullptr},  // sa3_w2
    {17, 1024, 512, 1024,  512, nullptr},  // sa3_w3
    {19, 512, 1024,  512, 1024, nullptr},  // fc1_w
    {21, 256,  512,  256,  512, nullptr},  // fc2_w
    {23,  40,  256,   64,  256, nullptr},  // fc3_w (rows padded 40->64)
  };
  for (int i = 0; i < 12; ++i)
    wd[i].wp = (_Float16*)alloc((size_t)wd[i].CoutPad * wd[i].CinPad * 2);

  int*       idx1 = (int*)alloc(32L * 512 * 32 * 4);
  int*       idx2 = (int*)alloc(32L * 128 * 64 * 4);
  _Float16*  XA   = (_Float16*)alloc(262144L * 160 * 2);   // 84 MB ping
  _Float16*  XB   = (_Float16*)alloc(524288L * 128 * 2);   // 134 MB pong
  _Float16*  p1   = (_Float16*)alloc(16384L * 128 * 2);
  _Float16*  p2   = (_Float16*)alloc(4096L * 256 * 2);
  // FC activations padded to 64 columns (cols 32..63 computed, never stored)
  _Float16*  g    = (_Float16*)alloc(64L * 1024 * 2);
  _Float16*  hA   = (_Float16*)alloc(64L * 512 * 2);
  _Float16*  hB   = (_Float16*)alloc(64L * 256 * 2);

  // ----- weight conversion --------------------------------------------------
  for (int i = 0; i < 12; ++i) {
    long tot = (long)wd[i].CoutPad * wd[i].CinPad;
    pad_weights_kernel<<<dim3(cdiv(tot, 256)), dim3(256), 0, stream>>>(
        (const float*)d_in[wd[i].ii], wd[i].wp, wd[i].Cout, wd[i].Cin,
        wd[i].CoutPad, wd[i].CinPad);
  }

  // NcolsPad must be a multiple of 64; Ncols is the store bound.
  auto gemm = [&](const _Float16* W, const _Float16* X, const float* bias,
                  int Cout, int CoutPad, int CinPad, long Ncols, long NcolsPad,
                  _Float16* Yh, float* Yf, int ldy, int relu) {
    int  mt2n   = CoutPad / 32;
    long macros = (long)mt2n * (NcolsPad / 64);
    long blocks = cdiv(macros, 4);                 // 4 waves / 128-thread block
    wmma_gemm_kernel<<<dim3(blocks), dim3(128), 0, stream>>>(
        W, X, bias, Cout, CinPad, mt2n, macros, Ncols, Yh, Yf, ldy, relu);
  };

  // ===== SA1: N=4096 -> P=512, K=32, MLP 6->64->64->128 =====================
  knn_kernel<<<dim3(32 * 512), dim3(256), 4096 * 4 + 8 * 8, stream>>>(
      x, 512, 4096, 32, idx1);
  {
    long tot = 32L * 512 * 32;
    gather_sa1_kernel<<<dim3(cdiv(tot, 256)), dim3(256), 0, stream>>>(x, idx1, XA);
  }
  long n1 = 32L * 512 * 32;
  gemm(wd[0].wp, XA, (const float*)d_in[2],  64,  64,  32, n1, n1, XB, nullptr,  64, 1);
  gemm(wd[1].wp, XB, (const float*)d_in[4],  64,  64,  64, n1, n1, XA, nullptr,  64, 1);
  gemm(wd[2].wp, XA, (const float*)d_in[6], 128, 128,  64, n1, n1, XB, nullptr, 128, 0);
  {
    long tot = 16384L * 128;
    maxpool_k_kernel<<<dim3(cdiv(tot, 256)), dim3(256), 0, stream>>>(XB, p1, 32, 128, 16384);
  }

  // ===== SA2: 512 -> P=128, K=64, MLP 131->128->128->256 ====================
  knn_kernel<<<dim3(32 * 128), dim3(256), 512 * 4 + 8 * 8, stream>>>(
      x, 128, 512, 64, idx2);
  {
    long tot = 32L * 128 * 64;
    gather_sa2_kernel<<<dim3(cdiv(tot, 256)), dim3(256), 0, stream>>>(x, idx2, p1, XA);
  }
  long n2 = 32L * 128 * 64;
  gemm(wd[3].wp, XA, (const float*)d_in[8],  128, 128, 160, n2, n2, XB, nullptr, 128, 1);
  gemm(wd[4].wp, XB, (const float*)d_in[10], 128, 128, 128, n2, n2, XA, nullptr, 128, 1);
  gemm(wd[5].wp, XA, (const float*)d_in[12], 256, 256, 128, n2, n2, XB, nullptr, 256, 0);
  {
    long tot = 4096L * 256;
    maxpool_k_kernel<<<dim3(cdiv(tot, 256)), dim3(256), 0, stream>>>(XB, p2, 64, 256, 4096);
  }

  // ===== SA3: global, nsample=1 -> MLP 259->256->512->1024, pool over P =====
  gather_sa3_kernel<<<dim3(cdiv(32L * 128, 256)), dim3(256), 0, stream>>>(p2, XA);
  long n3 = 32L * 128;
  gemm(wd[6].wp, XA, (const float*)d_in[14],  256,  256, 288, n3, n3, XB, nullptr,  256, 1);
  gemm(wd[7].wp, XB, (const float*)d_in[16],  512,  512, 256, n3, n3, XA, nullptr,  512, 1);
  gemm(wd[8].wp, XA, (const float*)d_in[18], 1024, 1024, 512, n3, n3, XB, nullptr, 1024, 0);
  pool_g_kernel<<<dim3(cdiv(32L * 1024, 256)), dim3(256), 0, stream>>>(XB, g);

  // ===== FC head: 1024 -> 512 -> 256 -> 40 (f32 out) ========================
  gemm(wd[9].wp,  g,  (const float*)d_in[20], 512, 512, 1024, 32, 64, hA, nullptr, 512, 1);
  gemm(wd[10].wp, hA, (const float*)d_in[22], 256, 256,  512, 32, 64, hB, nullptr, 256, 1);
  gemm(wd[11].wp, hB, (const float*)d_in[24],  40,  64,  256, 32, 64, nullptr, (float*)d_out, 40, 0);
}
